// Attention2D_64484638982282
// MI455X (gfx1250) — matrix operand order repaired and checked by reading, not compiled, not run on hardware
//
#include <hip/hip_runtime.h>
#include <hip/hip_bf16.h>
#include <math.h>

typedef __attribute__((ext_vector_type(16))) _Float16 v16h;
typedef __attribute__((ext_vector_type(8)))  _Float16 v8h;
typedef __attribute__((ext_vector_type(4)))  _Float16 v4h;
typedef __attribute__((ext_vector_type(8)))  float    v8f;
typedef __attribute__((ext_vector_type(4)))  float    v4f;

#define SEQ_LEN 1024
#define D_MODEL 512
#define NH 8
#define HD 64
#define NB 8

// ---------------------------------------------------------------------------
// WMMA helpers (CDNA5 wave32, V_WMMA_F32_16X16X32_F16)
// A fragment (16-bit, 16x32): lane (m = t&15, hi = t>>4) holds
//   halves 0..7  -> K = k0 + 8*hi + j ; halves 8..15 -> K = k0 + 16 + 8*hi + j
// B fragment (16-bit, 32x16): lane (n = t&15, hi = t>>4) holds
//   same K order as A, of column n (PV GEMM: A and B both K = k0 + 16*hi + j)
// C/D (f32 16x16): acc[r] is (row = r + 8*(t>>4), col = t&15)
// ---------------------------------------------------------------------------
__device__ __forceinline__ v16h make_frag(const _Float16* p0, const _Float16* p1) {
  v8h lo = *(const v8h*)p0;
  v8h hi = *(const v8h*)p1;
  v16h r;
#pragma unroll
  for (int i = 0; i < 8; ++i) { r[i] = lo[i]; r[i + 8] = hi[i]; }
  return r;
}

__device__ __forceinline__ v8f wmma_f16(v16h a, v16h b, v8f c) {
  return __builtin_amdgcn_wmma_f32_16x16x32_f16(false, a, false, b, (short)0, c,
                                                false, false);
}

// ---------------------------------------------------------------------------
// Kernel 0a: f32 -> f16 contiguous convert (8 elems/thread, b128 in/out)
// ---------------------------------------------------------------------------
__global__ __launch_bounds__(256) void cvt_f16_kernel(
    const float* __restrict__ x, _Float16* __restrict__ y, int n) {
  int i8 = (blockIdx.x * 256 + threadIdx.x) * 8;
  if (i8 + 7 >= n) return;
  v4f a = *(const v4f*)(x + i8);
  v4f b = *(const v4f*)(x + i8 + 4);
  v8h o;
#pragma unroll
  for (int j = 0; j < 4; ++j) { o[j] = (_Float16)a[j]; o[j + 4] = (_Float16)b[j]; }
  *(v8h*)(y + i8) = o;
}

// ---------------------------------------------------------------------------
// Kernel 0b: weight transpose+convert: W[k][n] (f32, 512x512) -> Wt[n][k] (f16)
// ---------------------------------------------------------------------------
__global__ __launch_bounds__(256) void wt_transpose_kernel(
    const float* __restrict__ W, _Float16* __restrict__ Wt) {
  __shared__ _Float16 tile[64][72];  // padded rows (144 B, 8B-aligned)
  const int t = threadIdx.x;
  const int k0 = blockIdx.x * 64;
  const int n0 = blockIdx.y * 64;
#pragma unroll
  for (int i = 0; i < 16; ++i) {
    int e = t + i * 256;
    int kk = e >> 6, nn = e & 63;  // coalesced read along n
    tile[nn][kk] = (_Float16)W[(size_t)(k0 + kk) * D_MODEL + n0 + nn];
  }
  __syncthreads();
#pragma unroll
  for (int i = 0; i < 4; ++i) {
    int e = t + i * 256;
    int nn = e >> 4, kk4 = (e & 15) * 4;  // coalesced write along k
    *(v4h*)&Wt[(size_t)(n0 + nn) * D_MODEL + k0 + kk4] = *(const v4h*)&tile[nn][kk4];
  }
}

// ---------------------------------------------------------------------------
// Kernel 1: projection GEMM  C[8192x512] = Ah[8192x512] x Wt^T + bias
// Ah: f16 (M,K). Wt: f16 (N,K). Pure-register WMMA, no LDS.
// Block = 256 threads (8 waves); wave tile 16x64 (4 acc): A-frag reused x4.
// One-stage software pipeline: iteration k+1 fragments are loaded before the
// iteration-k WMMAs, so loads overlap matrix math within the wave.
// mode 0: store (B,H,L,D). mode 1: store (B,H,D,L).
// ---------------------------------------------------------------------------
__global__ __launch_bounds__(256) void proj_gemm_kernel(
    const _Float16* __restrict__ Ah, const _Float16* __restrict__ Wt,
    const float* __restrict__ bias, _Float16* __restrict__ out, int mode) {
  const int t = threadIdx.x;
  const int wv = t >> 5, lane = t & 31;
  const int n = lane & 15, hi = lane >> 4;
  const int m0 = blockIdx.x * 16;
  const int colBase = wv * 64;  // wave's 64-column strip

  const _Float16* arow = Ah + (size_t)(m0 + n) * D_MODEL + 8 * hi;
  const _Float16* wrow = Wt + (size_t)(colBase + n) * D_MODEL + 8 * hi;

  v8f acc[4] = {};
  // ---- prologue: load K-step 0 fragments ----
  v16h af = make_frag(arow, arow + 16);
  v16h bf[4];
#pragma unroll
  for (int j = 0; j < 4; ++j) {
    const _Float16* wr = wrow + (size_t)j * 16 * D_MODEL;
    bf[j] = make_frag(wr, wr + 16);
  }
  // ---- pipelined main loop ----
#pragma unroll 2
  for (int k0 = 32; k0 < D_MODEL; k0 += 32) {
    v16h af_n = make_frag(arow + k0, arow + k0 + 16);
    v16h bf_n[4];
#pragma unroll
    for (int j = 0; j < 4; ++j) {
      const _Float16* wr = wrow + (size_t)j * 16 * D_MODEL + k0;
      bf_n[j] = make_frag(wr, wr + 16);
    }
#pragma unroll
    for (int j = 0; j < 4; ++j) acc[j] = wmma_f16(af, bf[j], acc[j]);
    af = af_n;
#pragma unroll
    for (int j = 0; j < 4; ++j) bf[j] = bf_n[j];
  }
  // ---- epilogue: last K-step ----
#pragma unroll
  for (int j = 0; j < 4; ++j) acc[j] = wmma_f16(af, bf[j], acc[j]);

  const int rowM0 = m0 + 8 * hi;  // first of 8 consecutive rows for this lane
  const int bb = rowM0 >> 10, ll = rowM0 & 1023;
#pragma unroll
  for (int j = 0; j < 4; ++j) {
    const int colN = colBase + j * 16 + n;  // 0..511
    const float bval = bias[colN];
    const int h = colN >> 6, d = colN & 63;
    if (mode == 0) {
#pragma unroll
      for (int r = 0; r < 8; ++r) {
        size_t idx = (((size_t)(bb * NH + h) * SEQ_LEN) + (ll + r)) * HD + d;
        out[idx] = (_Float16)(acc[j][r] + bval);
      }
    } else {
      v8h o;
#pragma unroll
      for (int r = 0; r < 8; ++r) o[r] = (_Float16)(acc[j][r] + bval);
      *(v8h*)&out[(((size_t)(bb * NH + h) * HD) + d) * SEQ_LEN + ll] = o;
    }
  }
}

// ---------------------------------------------------------------------------
// Kernel 2: sigma = query @ Ws + bs, stored as (B,H,L,2) f32.
// ---------------------------------------------------------------------------
__global__ __launch_bounds__(256) void sigma_kernel(
    const float* __restrict__ Q, const float* __restrict__ Ws,
    const float* __restrict__ bs, float* __restrict__ sigma) {
  const int t = threadIdx.x;
  const int wv = t >> 5, lane = t & 31;
  const int row = blockIdx.x * 8 + wv;  // 0..8191
  const int col = lane & 15, half = lane >> 4;
  const float* q = Q + (size_t)row * D_MODEL;
  float acc = 0.f;
  for (int j = half * 256; j < half * 256 + 256; ++j)
    acc += q[j] * Ws[j * (2 * NH) + col];
  acc += __shfl_xor(acc, 16, 32);
  if (half == 0) {
    int b = row >> 10, l = row & 1023;
    int h = col >> 1, comp = col & 1;
    sigma[(((size_t)(b * NH + h) * SEQ_LEN) + l) * 2 + comp] = acc + bs[col];
  }
}

// ---------------------------------------------------------------------------
// Kernel 3: fused attention per (b, h, 16-row tile).
// Scores in registers, softmax via shfl + small LDS reduce, P -> f16 LDS,
// PV GEMM with 8 waves (4 d-tiles x 2 K-halves, LDS combine).
// ---------------------------------------------------------------------------
__global__ __launch_bounds__(256) void attn_kernel(
    const _Float16* __restrict__ Qh, const _Float16* __restrict__ Kh,
    const _Float16* __restrict__ Vh, const float* __restrict__ ab,
    const int* __restrict__ bias_idxs, const float* __restrict__ scale_p,
    float* __restrict__ attn_out, float* __restrict__ out) {
  __shared__ _Float16 p16[16][SEQ_LEN];  // 32 KB normalized probs
  __shared__ float redmax[16][8];
  __shared__ float redsum[16][8];
  __shared__ float part[4][16][16];      // PV partial accumulators

  const int t = threadIdx.x;
  const int wv = t >> 5, lane = t & 31;
  const int n = lane & 15, hi = lane >> 4;
  const int lt = blockIdx.x, h = blockIdx.y, b = blockIdx.z;
  const int l0 = lt * 16;
  const float scl = scale_p[0];

  const _Float16* Qbase = Qh + (((size_t)(b * NH + h)) * SEQ_LEN + l0) * HD;
  const _Float16* Kbase = Kh + ((size_t)(b * NH + h)) * SEQ_LEN * HD;
  const _Float16* Vbase = Vh + ((size_t)(b * NH + h)) * HD * SEQ_LEN;

  const _Float16* qrow = Qbase + (size_t)n * HD;
  v16h aq0 = make_frag(qrow + 8 * hi, qrow + 16 + 8 * hi);
  v16h aq1 = make_frag(qrow + 32 + 8 * hi, qrow + 48 + 8 * hi);

  // ---- scores: each wave covers 8 s-tiles, kept in registers -------------
  float s[8][8];  // s[i][r]: row = r + 8*hi, col = (wv + 8*i)*16 + n
#pragma unroll
  for (int i = 0; i < 8; ++i) {
    const int s0 = (wv + 8 * i) * 16;
    const _Float16* krow = Kbase + (size_t)(s0 + n) * HD;
    v16h bk0 = make_frag(krow + 8 * hi, krow + 16 + 8 * hi);
    v16h bk1 = make_frag(krow + 32 + 8 * hi, krow + 48 + 8 * hi);
    v8f accs = {};
    accs = wmma_f16(aq0, bk0, accs);
    accs = wmma_f16(aq1, bk1, accs);
    const int col = s0 + n;
#pragma unroll
    for (int r = 0; r < 8; ++r) {
      const int row = l0 + r + 8 * hi;
      const int idx = bias_idxs[row * SEQ_LEN + col];
      s[i][r] = scl * accs[r] + ab[h * SEQ_LEN + idx];
    }
  }

  // ---- softmax: row max --------------------------------------------------
  float m8[8];
#pragma unroll
  for (int r = 0; r < 8; ++r) {
    float m = s[0][r];
#pragma unroll
    for (int i = 1; i < 8; ++i) m = fmaxf(m, s[i][r]);
    m = fmaxf(m, __shfl_xor(m, 1, 32));
    m = fmaxf(m, __shfl_xor(m, 2, 32));
    m = fmaxf(m, __shfl_xor(m, 4, 32));
    m = fmaxf(m, __shfl_xor(m, 8, 32));
    m8[r] = m;
  }
  if (n == 0) {
#pragma unroll
    for (int r = 0; r < 8; ++r) redmax[r + 8 * hi][wv] = m8[r];
  }
  __syncthreads();
  float mx[8];
#pragma unroll
  for (int r = 0; r < 8; ++r) {
    float m = redmax[r + 8 * hi][0];
#pragma unroll
    for (int w = 1; w < 8; ++w) m = fmaxf(m, redmax[r + 8 * hi][w]);
    mx[r] = m;
  }
  // ---- exp + row sum -----------------------------------------------------
  float s8[8];
#pragma unroll
  for (int r = 0; r < 8; ++r) {
    float sum = 0.f;
#pragma unroll
    for (int i = 0; i < 8; ++i) {
      float e = __expf(s[i][r] - mx[r]);
      s[i][r] = e;
      sum += e;
    }
    sum += __shfl_xor(sum, 1, 32);
    sum += __shfl_xor(sum, 2, 32);
    sum += __shfl_xor(sum, 4, 32);
    sum += __shfl_xor(sum, 8, 32);
    s8[r] = sum;
  }
  if (n == 0) {
#pragma unroll
    for (int r = 0; r < 8; ++r) redsum[r + 8 * hi][wv] = s8[r];
  }
  __syncthreads();
  float inv[8];
#pragma unroll
  for (int r = 0; r < 8; ++r) {
    float sum = 0.f;
#pragma unroll
    for (int w = 0; w < 8; ++w) sum += redsum[r + 8 * hi][w];
    inv[r] = 1.0f / sum;
  }
  // ---- normalize, write attn (f32) and P (f16 LDS) -----------------------
#pragma unroll
  for (int i = 0; i < 8; ++i) {
    const int col = (wv + 8 * i) * 16 + n;
#pragma unroll
    for (int r = 0; r < 8; ++r) {
      const int row = r + 8 * hi;
      const float p = s[i][r] * inv[r];
      attn_out[(((size_t)(b * NH + h) * SEQ_LEN) + (l0 + row)) * SEQ_LEN + col] = p;
      p16[row][col] = (_Float16)p;
    }
  }
  __syncthreads();

  // ---- out tile = P(16x1024) @ V(1024x64) --------------------------------
  const int dtile = wv & 3, khalf = wv >> 2;
  v8f acco = {};
#pragma unroll 2
  for (int ks = khalf * 512; ks < khalf * 512 + 512; ks += 32) {
    v16h ap = make_frag(&p16[n][ks + 16 * hi], &p16[n][ks + 16 * hi + 8]);
    const _Float16* vcol =
        Vbase + (size_t)(dtile * 16 + n) * SEQ_LEN + ks + 16 * hi;
    v16h bv = make_frag(vcol, vcol + 8);
    acco = wmma_f16(ap, bv, acco);
  }
  if (khalf == 1) {
#pragma unroll
    for (int r = 0; r < 8; ++r) part[dtile][r + 8 * hi][n] = acco[r];
  }
  __syncthreads();
  if (khalf == 0) {
#pragma unroll
    for (int r = 0; r < 8; ++r) {
      const int row = r + 8 * hi;
      const float v = acco[r] + part[dtile][row][n];
      out[((size_t)b * SEQ_LEN + (l0 + row)) * D_MODEL + h * HD + dtile * 16 + n] = v;
    }
  }
}

// ---------------------------------------------------------------------------
// Kernel 4: target map. One block per (b,h,l) row; 1024 cols = 256 x float4.
// ---------------------------------------------------------------------------
__global__ __launch_bounds__(256) void target_kernel(
    const float* __restrict__ sigma, const float* __restrict__ dist_x,
    const float* __restrict__ dist_y, float* __restrict__ target) {
  const int l = blockIdx.x, h = blockIdx.y, b = blockIdx.z;
  const size_t base = ((size_t)(b * NH + h) * SEQ_LEN + l);
  const float s1r = sigma[base * 2 + 0];
  const float s2r = sigma[base * 2 + 1];
  // s = 3^(sigmoid(5x)+1e-5) - 1
  const float g1 = 1.0f / (1.0f + __expf(-5.0f * s1r)) + 1e-5f;
  const float g2 = 1.0f / (1.0f + __expf(-5.0f * s2r)) + 1e-5f;
  const float s1 = __expf(g1 * 1.0986122886681098f) - 1.0f;
  const float s2 = __expf(g2 * 1.0986122886681098f) - 1.0f;
  const float c0 = 1.0f / (2.0f * 3.14159265358979323846f * s1 * s2);
  const float a1 = 0.5f / (s1 * s1);
  const float a2 = 0.5f / (s2 * s2);
  const int c = threadIdx.x * 4;
  v4f dy = *(const v4f*)(dist_y + (size_t)l * SEQ_LEN + c);
  v4f dx = *(const v4f*)(dist_x + (size_t)l * SEQ_LEN + c);
  v4f o;
#pragma unroll
  for (int j = 0; j < 4; ++j) o[j] = c0 * __expf(-(dy[j] * a1 + dx[j] * a2));
  *(v4f*)(target + base * SEQ_LEN + c) = o;
}

// ---------------------------------------------------------------------------
extern "C" void kernel_launch(void* const* d_in, const int* in_sizes, int n_in,
                              void* d_out, int out_size, void* d_ws,
                              size_t ws_size, hipStream_t stream) {
  const float* query = (const float*)d_in[0];
  const float* key   = (const float*)d_in[1];
  const float* value = (const float*)d_in[2];
  const float* Wq = (const float*)d_in[3];
  const float* bq = (const float*)d_in[4];
  const float* Wk = (const float*)d_in[5];
  const float* bk = (const float*)d_in[6];
  const float* Wv = (const float*)d_in[7];
  const float* bv = (const float*)d_in[8];
  const float* Ws = (const float*)d_in[9];
  const float* bs = (const float*)d_in[10];
  const float* scale = (const float*)d_in[11];
  const float* ab = (const float*)d_in[12];
  const float* dist_x = (const float*)d_in[13];
  const float* dist_y = (const float*)d_in[14];
  const int* bias_idxs = (const int*)d_in[15];

  float* out = (float*)d_out;                                  // (B,L,512)
  float* attn = out + (size_t)NB * SEQ_LEN * D_MODEL;          // (B,H,L,S)
  float* target = attn + (size_t)NB * NH * SEQ_LEN * SEQ_LEN;  // (B,H,L,S)

  char* ws = (char*)d_ws;
  _Float16* Qh = (_Float16*)ws;                                 // 8 MB (B,H,L,D)
  _Float16* Kh = (_Float16*)(ws + (size_t)8 * 1024 * 1024);     // 8 MB (B,H,S,D)
  _Float16* Vh = (_Float16*)(ws + (size_t)16 * 1024 * 1024);    // 8 MB (B,H,D,S)
  float* sigma = (float*)(ws + (size_t)24 * 1024 * 1024);       // 0.5 MB
  _Float16* Xh = (_Float16*)(ws + (size_t)25 * 1024 * 1024);    // 8 MB (reused)
  _Float16* WT = (_Float16*)(ws + (size_t)33 * 1024 * 1024);    // 0.5 MB (reused)

  const int nAct = NB * SEQ_LEN * D_MODEL;  // 4M elements
  dim3 gcvt(nAct / (256 * 8));
  dim3 gwt(D_MODEL / 64, D_MODEL / 64);
  dim3 gproj(512);

  // Q projection
  cvt_f16_kernel<<<gcvt, 256, 0, stream>>>(query, Xh, nAct);
  wt_transpose_kernel<<<gwt, 256, 0, stream>>>(Wq, WT);
  proj_gemm_kernel<<<gproj, 256, 0, stream>>>(Xh, WT, bq, Qh, 0);
  // K projection
  cvt_f16_kernel<<<gcvt, 256, 0, stream>>>(key, Xh, nAct);
  wt_transpose_kernel<<<gwt, 256, 0, stream>>>(Wk, WT);
  proj_gemm_kernel<<<gproj, 256, 0, stream>>>(Xh, WT, bk, Kh, 0);
  // V projection (output transposed to (B,H,D,L))
  cvt_f16_kernel<<<gcvt, 256, 0, stream>>>(value, Xh, nAct);
  wt_transpose_kernel<<<gwt, 256, 0, stream>>>(Wv, WT);
  proj_gemm_kernel<<<gproj, 256, 0, stream>>>(Xh, WT, bv, Vh, 1);

  sigma_kernel<<<1024, 256, 0, stream>>>(query, Ws, bs, sigma);
  attn_kernel<<<dim3(SEQ_LEN / 16, NH, NB), 256, 0, stream>>>(
      Qh, Kh, Vh, ab, bias_idxs, scale, attn, out);
  target_kernel<<<dim3(SEQ_LEN, NH, NB), 256, 0, stream>>>(sigma, dist_x,
                                                           dist_y, target);
}
